// ScaledDotProductAttention_20564303413275
// MI455X (gfx1250) — matrix operand order repaired and checked by reading, not compiled, not run on hardware
//
#include <hip/hip_runtime.h>

#define B_DIM 8
#define T_DIM 2048
#define D_DIM 1024
#define JSTEP 32
#define NTHREADS 512        // 16 waves: 2 query-row groups x 8 D-chunk waves

typedef __attribute__((ext_vector_type(16))) __bf16 v16bf;
typedef __attribute__((ext_vector_type(8)))  float  v8f;

union BF16Frag {
  v16bf v;
  unsigned short u[16];
  uint4 q[2];
};

__device__ __forceinline__ unsigned short f2bf(float f) {
  union { __bf16 h; unsigned short u; } cv;
  cv.h = (__bf16)f;                       // native bf16 convert (pairable)
  return cv.u;
}

// One workgroup (16 waves, 512 threads) handles a 32-row query tile.
// Row-group g = wave>>3 (rows g*16..g*16+15); wave&7 selects the 128-wide D-chunk.
// The 32-key V tile staging is shared by both row groups -> 2x compute per staged byte.
__global__ __launch_bounds__(NTHREADS, 1)
void attn_fwd_kernel(const float* __restrict__ Q,   // "key" input, acts as queries
                     const float* __restrict__ V,   // "value" input, acts as keys AND values
                     float* __restrict__ O) {
  __shared__ unsigned short Vt[D_DIM][JSTEP];     // transposed bf16 V tile (P*V B-ops): 64 KB
  __shared__ unsigned short Vrow[JSTEP][D_DIM];   // row-major bf16 V tile (score B-ops): 64 KB
  __shared__ float          Sred[2][16][JSTEP];   // per row-group score reduction: 4 KB
  __shared__ unsigned short Pbf[2][16][JSTEP];    // softmax probs (bf16): 2 KB
  __shared__ float          alphaS[2][16];
  __shared__ float          mS[2][16];
  __shared__ float          lS[2][16];

  const int tid   = threadIdx.x;
  const int wave  = tid >> 5;
  const int lane  = tid & 31;
  const int laneN = lane & 15;
  const int hi    = (lane >> 4) & 1;     // 0: lanes 0-15, 1: lanes 16-31
  const int g     = wave >> 3;           // query row-group 0/1
  const int dbase = (wave & 7) * 128;    // this wave's D-chunk

  const int b    = blockIdx.x >> 6;      // T/32 = 64 row tiles per batch
  const int it   = blockIdx.x & 63;
  const int row0 = it * 32;

  const float scale = 0.03125f;          // 1/sqrt(1024)

  // softmax distribution: 1024 elements / 512 threads = 2 each
  const int sg   = tid >> 8;             // row group
  const int srow = (tid >> 4) & 15;
  const int scol = (tid & 15) * 2;

  if (tid < 32) { mS[tid >> 4][tid & 15] = -1e30f; lS[tid >> 4][tid & 15] = 0.f; }

  // ---- Load Q tile once: 4 A-fragments (16x32 bf16) per wave ----
  BF16Frag qfrag[4];
  {
    const size_t qrow = (size_t)(b * T_DIM + row0 + g * 16 + laneN) * D_DIM;
    const int klo = hi ? 8 : 0;          // A layout: lanes<16 hold K{0-7,16-23}, lanes>=16 K{8-15,24-31}
    #pragma unroll
    for (int c = 0; c < 4; ++c) {
      const float* p = Q + qrow + dbase + c * 32;
      float f[16];
      *(float4*)(f + 0)  = *(const float4*)(p + klo);
      *(float4*)(f + 4)  = *(const float4*)(p + klo + 4);
      *(float4*)(f + 8)  = *(const float4*)(p + 16 + klo);
      *(float4*)(f + 12) = *(const float4*)(p + 16 + klo + 4);
      #pragma unroll
      for (int e = 0; e < 16; ++e) qfrag[c].u[e] = f2bf(f[e]);
    }
  }

  v8f acc[8];
  #pragma unroll
  for (int t = 0; t < 8; ++t) acc[t] = (v8f)0.f;

  const int stgK  = tid >> 4;            // staging: key row 0..31
  const int stgDG = tid & 15;            // staging: d-group 0..15

  for (int j = 0; j < T_DIM; j += JSTEP) {
    __syncthreads();                     // prev iter's LDS tiles fully consumed

    // zero score reduction buffers (1024 floats / 512 threads)
    ((float*)Sred)[tid]            = 0.f;
    ((float*)Sred)[tid + NTHREADS] = 0.f;

    // ---- Stage bf16 V tile in BOTH layouts (single pass over global) ----
    {
      const float* gsrc = V + (size_t)(b * T_DIM + j + stgK) * D_DIM;
      #pragma unroll 4
      for (int c = 0; c < 16; ++c) {
        const int d0 = stgDG * 4 + c * 64;
        float4 f = *(const float4*)(gsrc + d0);
        const unsigned short h0 = f2bf(f.x), h1 = f2bf(f.y);
        const unsigned short h2 = f2bf(f.z), h3 = f2bf(f.w);
        Vt[d0 + 0][stgK] = h0;
        Vt[d0 + 1][stgK] = h1;
        Vt[d0 + 2][stgK] = h2;
        Vt[d0 + 3][stgK] = h3;
        uint2 pk;
        pk.x = (unsigned)h0 | ((unsigned)h1 << 16);
        pk.y = (unsigned)h2 | ((unsigned)h3 << 16);
        *(uint2*)&Vrow[stgK][d0] = pk;
      }
      // prefetch next j-tile (1024 cachelines / 512 threads = 2 each)
      if (j + JSTEP < T_DIM) {
        const float* nxt = V + (size_t)(b * T_DIM + j + JSTEP + stgK) * D_DIM + stgDG * 64;
        __builtin_prefetch(nxt, 0, 1);
        __builtin_prefetch(nxt + 32, 0, 1);
      }
    }

    __syncthreads();                     // Sred zeroed + tiles staged

    // ---- Partial scores from LDS row-major tile (wave's 128-wide D-chunk) ----
    #pragma unroll
    for (int h = 0; h < 2; ++h) {
      v8f s = (v8f)0.f;
      const unsigned short* vr = &Vrow[h * 16 + laneN][0];
      const int klo = hi ? 8 : 0;        // B layout: lanes<16 hold K{0-7,16-23}, lanes>=16 K{8-15,24-31}
      #pragma unroll
      for (int c = 0; c < 4; ++c) {
        BF16Frag kb;
        kb.q[0] = *(const uint4*)(vr + dbase + c * 32 + klo);
        kb.q[1] = *(const uint4*)(vr + dbase + c * 32 + klo + 16);
        s = __builtin_amdgcn_wmma_f32_16x16x32_bf16(false, qfrag[c].v, false, kb.v,
                                                    (short)0, s, false, false);
      }
      #pragma unroll
      for (int r = 0; r < 8; ++r)
        atomicAdd(&Sred[g][r + hi * 8][h * 16 + laneN], s[r]);   // ds_add_f32
    }

    __syncthreads();

    // ---- Online softmax: fully parallel (2 elements/thread, shfl row-reduce) ----
    {
      const float s0 = Sred[sg][srow][scol]     * scale;
      const float s1 = Sred[sg][srow][scol + 1] * scale;
      const float m_old = mS[sg][srow];
      float mx = fmaxf(s0, s1);
      #pragma unroll
      for (int off = 1; off < 16; off <<= 1)
        mx = fmaxf(mx, __shfl_xor(mx, off, 32));   // stays within 16-lane half
      mx = fmaxf(mx, m_old);
      const float p0 = __expf(s0 - mx);
      const float p1 = __expf(s1 - mx);
      float sum = p0 + p1;
      #pragma unroll
      for (int off = 1; off < 16; off <<= 1)
        sum += __shfl_xor(sum, off, 32);
      unsigned int pk = (unsigned)f2bf(p0) | ((unsigned)f2bf(p1) << 16);
      *(unsigned int*)&Pbf[sg][srow][scol] = pk;
      if ((tid & 15) == 0) {
        const float alpha = __expf(m_old - mx);
        alphaS[sg][srow] = alpha;
        lS[sg][srow]     = lS[sg][srow] * alpha + sum;
        mS[sg][srow]     = mx;
      }
    }

    __syncthreads();

    // ---- Rescale accumulators by alpha[row] ----
    float a_r[8];
    #pragma unroll
    for (int r = 0; r < 8; ++r) a_r[r] = alphaS[g][r + hi * 8];
    #pragma unroll
    for (int t = 0; t < 8; ++t)
      #pragma unroll
      for (int r = 0; r < 8; ++r) acc[t][r] *= a_r[r];

    // ---- P (16x32 bf16) as A operand from LDS ----
    BF16Frag pa;
    {
      const int klo = hi ? 8 : 0;
      pa.q[0] = *(const uint4*)&Pbf[g][laneN][klo];
      pa.q[1] = *(const uint4*)&Pbf[g][laneN][klo + 16];
    }

    // ---- O_chunk += P x V_tile : 8 WMMAs over this wave's D columns ----
    #pragma unroll
    for (int t = 0; t < 8; ++t) {
      BF16Frag vb;
      const int d   = dbase + t * 16 + laneN;   // B column n = output d-column
      const int klo = hi ? 8 : 0;
      vb.q[0] = *(const uint4*)&Vt[d][klo];
      vb.q[1] = *(const uint4*)&Vt[d][klo + 16];
      acc[t] = __builtin_amdgcn_wmma_f32_16x16x32_bf16(false, pa.v, false, vb.v,
                                                       (short)0, acc[t], false, false);
    }
  }

  // lS final after last softmax barrier; normalize and store
  float invl[8];
  #pragma unroll
  for (int r = 0; r < 8; ++r) invl[r] = 1.f / lS[g][r + hi * 8];

  #pragma unroll
  for (int t = 0; t < 8; ++t) {
    const int col = dbase + t * 16 + laneN;
    #pragma unroll
    for (int r = 0; r < 8; ++r) {
      const int row = row0 + g * 16 + r + hi * 8;
      O[((size_t)(b * T_DIM) + row) * D_DIM + col] = acc[t][r] * invl[r];
    }
  }
}

extern "C" void kernel_launch(void* const* d_in, const int* in_sizes, int n_in,
                              void* d_out, int out_size, void* d_ws, size_t ws_size,
                              hipStream_t stream) {
  (void)in_sizes; (void)n_in; (void)out_size; (void)d_ws; (void)ws_size;
  const float* key   = (const float*)d_in[0];   // acts as queries
  const float* value = (const float*)d_in[1];   // acts as keys and values
  float* out = (float*)d_out;

  dim3 grid(B_DIM * (T_DIM / 32));   // 512 workgroups
  dim3 block(NTHREADS);              // 16 waves (wave32)
  attn_fwd_kernel<<<grid, block, 0, stream>>>(key, value, out);
}
